// ConvCapsuleLayer_8753143349816
// MI455X (gfx1250) — matrix operand order repaired and checked by reading, not compiled, not run on hardware
//
#include <hip/hip_runtime.h>
#include <stdint.h>

typedef __attribute__((ext_vector_type(16))) __bf16 v16bf;
typedef __attribute__((ext_vector_type(8)))  float  v8f;

#define NCAP   8     // output capsules
#define NATOM  16    // output atoms
#define ICAPS  4     // input capsules
#define ATOMS  16    // input atoms
#define DDIM   24
#define HDIM   48
#define WDIM   48
#define BDIM   2
#define KSTEPS 14    // ceil(27 taps / 2), K=32 per step (2 taps x 16 atoms)

__device__ __forceinline__ uint32_t f2bf1(float f) {
    uint32_t u = __float_as_uint(f);
    return (u + 0x7FFFu + ((u >> 16) & 1u)) >> 16;   // round-to-nearest-even
}
__device__ __forceinline__ uint32_t packbf(float lo, float hi) {
    return f2bf1(lo) | (f2bf1(hi) << 16);
}

// ---------------------------------------------------------------------------
// Kernel 1: pack conv weights into per-lane WMMA B fragments in a contiguous K order (same as A).
// B 32x16 in this code's order: lanes 0-15 hold K=0..15 (VGPR r -> K=2r,2r+1), lanes 16-31
// hold K=16..31. K index = tap_local*16 + atom, tap = 2*ks + (lane>>4).
// Linear index: t = ((nc*KSTEPS + ks)*32 + lane)*8 + r
// ---------------------------------------------------------------------------
__global__ void pack_weights_kernel(const float* __restrict__ w,
                                    uint32_t* __restrict__ wB) {
    int t = blockIdx.x * blockDim.x + threadIdx.x;
    if (t >= NCAP * KSTEPS * 32 * 8) return;
    int r    = t & 7;
    int lane = (t >> 3) & 31;
    int ks   = (t >> 8) % KSTEPS;
    int nc   = t / (KSTEPS * 256);
    int na   = lane & 15;
    int tap  = 2 * ks + (lane >> 4);
    uint32_t val = 0u;
    if (tap < 27) {
        int a0 = 2 * r;
        // conv_w shape [128,16,3,3,3]; o = nc*16+na, strides: o:432, a:27, tap:1
        float f0 = w[((nc * 16 + na) * 16 + a0    ) * 27 + tap];
        float f1 = w[((nc * 16 + na) * 16 + a0 + 1) * 27 + tap];
        val = packbf(f0, f1);
    }
    wB[t] = val;
}

// ---------------------------------------------------------------------------
// Kernel 2: fused conv (implicit GEMM via bf16 WMMA) + full dynamic routing.
// One WG = 16 consecutive W positions at fixed (b,d,h). 8 waves, wave = nc.
// ---------------------------------------------------------------------------
#define XTILE_DW (3 * 3 * 18 * ICAPS * 8)   // 5184 dwords (bf16 atom pairs)

__launch_bounds__(256)
__global__ void capsconv_route_kernel(const float* __restrict__ x,
                                      const uint32_t* __restrict__ wB,
                                      const float* __restrict__ bias,
                                      float* __restrict__ out) {
    __shared__ uint32_t lds_x[XTILE_DW];            // 20736 B
    __shared__ float    lds_logits[ICAPS * 16 * NCAP]; // 2048 B

    const int tile = blockIdx.x;
    const int wt = tile % 3;
    const int h  = (tile / 3) % HDIM;
    const int d  = (tile / (3 * HDIM)) % DDIM;
    const int bb = tile / (3 * HDIM * DDIM);
    const int w0 = wt * 16;

    const int tid  = threadIdx.x;
    const int lane = tid & 31;
    const int nc   = tid >> 5;        // wave id == output capsule
    const int na   = lane & 15;
    const int hi   = lane >> 4;       // half-wave: selects rows m / m+8

    // ---- stage zero-padded x halo tile into LDS as packed bf16 pairs ----
    // layout: ((ddl*3 + hhl)*18 + wwl)*4 + ic, 8 atom-pair dwords each
    for (int t = tid; t < XTILE_DW; t += 256) {
        int apair = t & 7;
        int ic    = (t >> 3) & 3;
        int wwl   = (t >> 5) % 18;
        int hhl   = (t / 576) % 3;
        int ddl   = t / 1728;
        int dg = d + ddl - 1, hg = h + hhl - 1, wg = w0 + wwl - 1;
        uint32_t v = 0u;
        if ((unsigned)dg < DDIM && (unsigned)hg < HDIM && (unsigned)wg < WDIM) {
            const float* px = x +
                (((((size_t)bb * DDIM + dg) * HDIM + hg) * WDIM + wg) * ICAPS + ic)
                    * ATOMS + apair * 2;
            v = packbf(px[0], px[1]);
        }
        lds_x[t] = v;
    }
    __syncthreads();

    // ---- implicit-GEMM conv: votes[ic] (16 rows x 16 na) per wave ----
    v8f vacc[ICAPS];
    #pragma unroll
    for (int ic = 0; ic < ICAPS; ++ic) { v8f z = {}; vacc[ic] = z; }

    const uint32_t* bpn = wB + (size_t)nc * (KSTEPS * 256) + lane * 8;

    #pragma unroll
    for (int ks = 0; ks < KSTEPS; ++ks) {
        // B fragment: 8 dwords, contiguous 32B per lane -> 2x global_load_b128
        union { uint32_t u[8]; v16bf v; } Bf;
        const uint32_t* bpk = bpn + ks * 256;
        uint4 b0 = *(const uint4*)(bpk);
        uint4 b1 = *(const uint4*)(bpk + 4);
        Bf.u[0] = b0.x; Bf.u[1] = b0.y; Bf.u[2] = b0.z; Bf.u[3] = b0.w;
        Bf.u[4] = b1.x; Bf.u[5] = b1.y; Bf.u[6] = b1.z; Bf.u[7] = b1.w;

        #pragma unroll
        for (int ic = 0; ic < ICAPS; ++ic) {
            // A fragment, contiguous K order matching B:
            // lane l, vgpr r: K = (l/16)*16 + 2*r {,+1}
            union { uint32_t u[8]; v16bf v; } Af;
            #pragma unroll
            for (int r = 0; r < 8; ++r) {
                int tap = 2 * ks + hi;
                uint32_t v = 0u;
                if (tap < 27) {
                    int kd = tap / 9, kh = (tap / 3) % 3, kw = tap % 3;
                    int wwl   = (lane & 15) + kw;   // row m = lane&15, +halo
                int apair = r;
                    v = lds_x[(((kd * 3 + kh) * 18 + wwl) * ICAPS + ic) * 8 + apair];
                }
                Af.u[r] = v;
            }
            vacc[ic] = __builtin_amdgcn_wmma_f32_16x16x32_bf16(
                false, Af.v, false, Bf.v, (short)0, vacc[ic], false, false);
        }
    }

    // ---- dynamic routing, fully in-register (votes never touch HBM) ----
    const float bias_v = bias[nc * NATOM + na];
    float logit[ICAPS][8];
    #pragma unroll
    for (int ic = 0; ic < ICAPS; ++ic)
        #pragma unroll
        for (int r = 0; r < 8; ++r) logit[ic][r] = 0.0f;

    float actv[8];
    for (int it = 0; it < 3; ++it) {
        // publish logits (replicated across na-group; one lane per group writes)
        if ((lane & 15) == 0) {
            #pragma unroll
            for (int ic = 0; ic < ICAPS; ++ic)
                #pragma unroll
                for (int r = 0; r < 8; ++r) {
                    int m = r + 8 * hi;
                    lds_logits[(ic * 16 + m) * NCAP + nc] = logit[ic][r];
                }
        }
        __syncthreads();

        float pre[8];
        #pragma unroll
        for (int r = 0; r < 8; ++r) pre[r] = bias_v;

        #pragma unroll
        for (int ic = 0; ic < ICAPS; ++ic) {
            #pragma unroll
            for (int r = 0; r < 8; ++r) {
                int m = r + 8 * hi;
                const float* lp = &lds_logits[(ic * 16 + m) * NCAP];
                float mx = lp[0];
                #pragma unroll
                for (int j = 1; j < NCAP; ++j) mx = fmaxf(mx, lp[j]);
                float s = 0.0f;
                #pragma unroll
                for (int j = 0; j < NCAP; ++j) s += __expf(lp[j] - mx);
                float route = __expf(logit[ic][r] - mx) / s;
                pre[r] += route * vacc[ic][r];
            }
        }
        // squash: norm2 over na = butterfly reduce within 16-lane group
        #pragma unroll
        for (int r = 0; r < 8; ++r) {
            float n2 = pre[r] * pre[r];
            n2 += __shfl_xor(n2, 1); n2 += __shfl_xor(n2, 2);
            n2 += __shfl_xor(n2, 4); n2 += __shfl_xor(n2, 8);
            actv[r] = pre[r] * (__fsqrt_rn(n2) / (1.0f + n2));
        }
        __syncthreads();   // reads of lds_logits done before next publish
        if (it < 2) {
            #pragma unroll
            for (int ic = 0; ic < ICAPS; ++ic)
                #pragma unroll
                for (int r = 0; r < 8; ++r) {
                    float dv = vacc[ic][r] * actv[r];
                    dv += __shfl_xor(dv, 1); dv += __shfl_xor(dv, 2);
                    dv += __shfl_xor(dv, 4); dv += __shfl_xor(dv, 8);
                    logit[ic][r] += dv;
                }
        }
    }

    // ---- store out[b,d,h,w,nc,na]; 16 lanes write 64B contiguous runs ----
    #pragma unroll
    for (int r = 0; r < 8; ++r) {
        int m = r + 8 * hi;
        size_t oidx = (((((size_t)bb * DDIM + d) * HDIM + h) * WDIM + (w0 + m))
                          * NCAP + nc) * NATOM + na;
        out[oidx] = actv[r];
    }
}

// ---------------------------------------------------------------------------
extern "C" void kernel_launch(void* const* d_in, const int* in_sizes, int n_in,
                              void* d_out, int out_size, void* d_ws, size_t ws_size,
                              hipStream_t stream) {
    const float* x  = (const float*)d_in[0];   // [2,24,48,48,4,16] f32
    const float* w  = (const float*)d_in[1];   // [128,16,3,3,3]    f32
    const float* b  = (const float*)d_in[2];   // [1,1,1,8,16]      f32
    float* out      = (float*)d_out;           // [2,24,48,48,8,16] f32
    uint32_t* wB    = (uint32_t*)d_ws;         // 28672 dwords = 112 KB

    int wElems = NCAP * KSTEPS * 32 * 8;       // 28672
    pack_weights_kernel<<<(wElems + 255) / 256, 256, 0, stream>>>(w, wB);

    int nTiles = BDIM * DDIM * HDIM * (WDIM / 16);  // 6912 workgroups
    capsconv_route_kernel<<<nTiles, 256, 0, stream>>>(x, wB, b, out);
}
